// CausalSelfAttention_43009802502282
// MI455X (gfx1250) — matrix operand order repaired and checked by reading, not compiled, not run on hardware
//
#include <hip/hip_runtime.h>

// ---------------------------------------------------------------------------
// Causal self-attention with external memory, MI455X (gfx1250, wave32).
// All matmuls on v_wmma_f32_16x16x32_f16 (f32 accumulate). The whole problem
// (~95MB) is L2-resident; compute-bound, so the structure keeps the matrix
// pipe fed: LDS-staged double-buffered B tiles, 32x64 per-wave output tiles
// (8 WMMAs per k-step per wave), batched B-fragment ds-loads, register-
// pipelined A fragments, early-issued V fragments in attention.
// __launch_bounds__(256,1): optimize for registers, not occupancy — no spills.
// ---------------------------------------------------------------------------

typedef _Float16 f16;
typedef _Float16 v8h  __attribute__((ext_vector_type(8)));
typedef _Float16 v16h __attribute__((ext_vector_type(16)));
typedef float    v8f  __attribute__((ext_vector_type(8)));

#define BB 4
#define TT 2048
#define CC 1024
#define HH 16
#define DD 64
#define MM 256
#define SS (MM + TT)   // 2304 keys per head
#define C3 (3 * CC)    // 3072

static __device__ __forceinline__ v8f wmma_f16(v16h a, v16h b, v8f c) {
  // D = A(16x32 f16) * B(32x16 f16) + C(16x16 f32)
  return __builtin_amdgcn_wmma_f32_16x16x32_f16(false, a, false, b,
                                                (short)0, c, false, false);
}

// A-fragment (16x32, f16) from a row-major matrix, row stride `ld` elements.
// Layout: lane L (row = L%16, khalf = L/16) holds K = khalf*16 + e
// (matches B fragments) -> two contiguous 8xf16 (16B) loads.
static __device__ __forceinline__ v16h load_afrag(const f16* base, int ld, int lane) {
  int row = lane & 15, kh = lane >> 4;
  const f16* p = base + (size_t)row * ld + kh * 16;
  v16h a;
  *((v8h*)&a)       = *(const v8h*)(p);
  *(((v8h*)&a) + 1) = *(const v8h*)(p + 8);
  return a;
}

// B-fragment (32x16, f16): lane holds column n = lane%16, elements are
// K = khalf*16 + e. Source stored column-major (Bt[n][k], row stride ld).
static __device__ __forceinline__ v16h load_bfrag(const f16* base, int ld, int lane) {
  int n = lane & 15, kh = lane >> 4;
  return *(const v16h*)(base + (size_t)n * ld + kh * 16);
}

// B-fragment from an LDS-staged [64][32] tile (row stride 32).
static __device__ __forceinline__ v16h load_bfrag_lds(const f16* base, int lane) {
  int n = lane & 15, kh = lane >> 4;
  return *(const v16h*)(base + n * 32 + kh * 16);
}

// Cooperative global->reg / reg->LDS staging of one 64(n) x 32(k) B tile.
// 256 threads x 8 f16 (16B) each.
static __device__ __forceinline__ v8h gload_btile(const f16* wt, int n0, int k0, int tid) {
  int nc = tid >> 2;           // 0..63
  int kk = (tid & 3) * 8;      // 0,8,16,24
  return *(const v8h*)(wt + (size_t)(n0 + nc) * CC + k0 + kk);
}
static __device__ __forceinline__ void sstore_btile(f16* bs, v8h v, int tid) {
  int nc = tid >> 2, kk = (tid & 3) * 8;
  *(v8h*)(bs + nc * 32 + kk) = v;
}

// ----------------------------- conversions ---------------------------------

__global__ void k_convert_x(const float* __restrict__ x, f16* __restrict__ xh, int n) {
  int i = blockIdx.x * blockDim.x + threadIdx.x;
  if (i < n) xh[i] = (f16)x[i];
}

// w: [K][N] f32 -> wt: [N][K] f16 (so GEMM B-fragments are contiguous)
__global__ void k_transpose_w(const float* __restrict__ w, f16* __restrict__ wt,
                              int K, int N) {
  int i = blockIdx.x * blockDim.x + threadIdx.x;
  if (i < K * N) {
    int nn = i / K, kk = i % K;
    wt[i] = (f16)w[(size_t)kk * N + nn];
  }
}

// ext_mem [B, 256, 1024] -> head-split into K rows [0,256) and V^T cols [0,256)
__global__ void k_mem_convert(const float* __restrict__ em,
                              f16* __restrict__ kh, f16* __restrict__ vt) {
  int i = blockIdx.x * blockDim.x + threadIdx.x;
  if (i >= BB * MM * CC) return;
  int b = i / (MM * CC);
  int r = i % (MM * CC);
  int m = r / CC, c = r % CC;
  int h = c / DD, d = c % DD;
  f16 v = (f16)em[i];
  int bh = b * HH + h;
  kh[((size_t)bh * SS + m) * DD + d] = v;      // K: [B,H,S,D]
  vt[((size_t)bh * DD + d) * SS + m] = v;      // V^T: [B,H,D,S]
}

// ------------------------------ GEMM core ----------------------------------
// Block = 8 waves, 256(M) x 64(N); each wave a 32x64 tile (8 WMMAs/k-step).
// B tile shared through LDS, double buffered; A fragments pipelined ahead;
// B fragments consumed pairwise so each dies after two WMMAs.

#define GEMM_CORE(ABASE, WT, N0, CACC)                                         \
  __shared__ f16 bs[2][64 * 32];                                               \
  const int NK = CC / 32;                                                      \
  v8h pre = gload_btile(WT, N0, 0, tid);                                       \
  sstore_btile(bs[0], pre, tid);                                               \
  pre = gload_btile(WT, N0, 32, tid);                                          \
  v16h ac0 = load_afrag(ABASE, CC, lane);                                      \
  v16h ac1 = load_afrag(ABASE + 16 * CC, CC, lane);                            \
  __syncthreads();                                                             \
  for (int ks = 0;;) {                                                         \
    v16h an0 = ac0, an1 = ac1;                                                 \
    if (ks + 1 < NK) {                                                         \
      an0 = load_afrag(ABASE + (ks + 1) * 32, CC, lane);                       \
      an1 = load_afrag(ABASE + 16 * CC + (ks + 1) * 32, CC, lane);             \
    }                                                                          \
    const f16* bb = bs[ks & 1];                                                \
    {                                                                          \
      v16h bf = load_bfrag_lds(bb + 0 * 512, lane);                            \
      CACC[0][0] = wmma_f16(ac0, bf, CACC[0][0]);                              \
      CACC[1][0] = wmma_f16(ac1, bf, CACC[1][0]);                              \
    }                                                                          \
    {                                                                          \
      v16h bf = load_bfrag_lds(bb + 1 * 512, lane);                            \
      CACC[0][1] = wmma_f16(ac0, bf, CACC[0][1]);                              \
      CACC[1][1] = wmma_f16(ac1, bf, CACC[1][1]);                              \
    }                                                                          \
    {                                                                          \
      v16h bf = load_bfrag_lds(bb + 2 * 512, lane);                            \
      CACC[0][2] = wmma_f16(ac0, bf, CACC[0][2]);                              \
      CACC[1][2] = wmma_f16(ac1, bf, CACC[1][2]);                              \
    }                                                                          \
    {                                                                          \
      v16h bf = load_bfrag_lds(bb + 3 * 512, lane);                            \
      CACC[0][3] = wmma_f16(ac0, bf, CACC[0][3]);                              \
      CACC[1][3] = wmma_f16(ac1, bf, CACC[1][3]);                              \
    }                                                                          \
    ++ks;                                                                      \
    if (ks == NK) break;                                                       \
    ac0 = an0; ac1 = an1;                                                      \
    __syncthreads();                      /* readers done with bs[ks&1] */     \
    sstore_btile(bs[ks & 1], pre, tid);   /* fill buffer for this iter */      \
    if (ks + 1 < NK) pre = gload_btile(WT, N0, (ks + 1) * 32, tid);            \
    __syncthreads();                      /* publish */                        \
  }

// ------------------------------ QKV GEMM -----------------------------------

__global__ void __launch_bounds__(256, 1)
k_gemm_qkv(const f16* __restrict__ xh, const f16* __restrict__ wt,
           const float* __restrict__ bias,
           f16* __restrict__ qh, f16* __restrict__ kh, f16* __restrict__ vt) {
  int tid = threadIdx.x;
  int lane = tid & 31;
  int wave = tid >> 5;
  int m0 = blockIdx.y * 256 + wave * 32;
  int n0 = blockIdx.x * 64;
  const f16* abase = xh + (size_t)m0 * CC;
  v8f c[2][4] = {};
  GEMM_CORE(abase, wt, n0, c)

  int half = lane >> 4, nl = lane & 15;
#pragma unroll
  for (int i = 0; i < 2; ++i) {
#pragma unroll
    for (int j = 0; j < 4; ++j) {
      int ncol = n0 + j * 16 + nl;
      float bv = bias[ncol];
      int sec = ncol / CC;        // 0=q, 1=k, 2=v
      int cc = ncol % CC;
      int h = cc / DD, d = cc % DD;
#pragma unroll
      for (int r = 0; r < 8; ++r) {
        int m = m0 + i * 16 + r + 8 * half;   // global row = b*T + t
        int b = m / TT, t = m % TT;
        int bh = b * HH + h;
        f16 val = (f16)(c[i][j][r] + bv);
        if (sec == 0)      qh[((size_t)bh * TT + t) * DD + d] = val;
        else if (sec == 1) kh[((size_t)bh * SS + (MM + t)) * DD + d] = val;
        else               vt[((size_t)bh * DD + d) * SS + (MM + t)] = val;
      }
    }
  }
}

// ---------------------------- flash attention ------------------------------
// One wave per (b, h, 16-row q tile). Keys processed 32 at a time:
// 4 WMMAs for scores, online softmax (shfl reductions within 16-lane halves),
// probs re-fragmented through per-wave LDS, 4 WMMAs for P@V. V fragments are
// issued BEFORE the softmax so the VALU work hides their latency.

__global__ void __launch_bounds__(256, 1)
k_attn(const f16* __restrict__ qh, const f16* __restrict__ kh,
       const f16* __restrict__ vt, f16* __restrict__ yh) {
  __shared__ f16 plds[8][16 * 32];   // per-wave 16x32 prob tile (8KB total)
  int lane = threadIdx.x & 31;
  int wavib = threadIdx.x >> 5;
  int wid = blockIdx.x * 8 + wavib;  // 8192 waves = B*H*(T/16)
  int qt = wid & 127;
  int h  = (wid >> 7) & 15;
  int b  = wid >> 11;
  int q0 = qt * 16;
  int bh = b * HH + h;
  const f16* qbase = qh + ((size_t)bh * TT + q0) * DD;
  const f16* kbase = kh + (size_t)bh * SS * DD;
  const f16* vbase = vt + (size_t)bh * DD * SS;

  v16h aq0 = load_afrag(qbase, DD, lane);        // d = 0..31
  v16h aq1 = load_afrag(qbase + 32, DD, lane);   // d = 32..63

  v8f o[4] = {};
  float mrow[8], lrow[8];
#pragma unroll
  for (int r = 0; r < 8; ++r) { mrow[r] = -3.0e38f; lrow[r] = 0.f; }

  int half = lane >> 4, nl = lane & 15;
  int smax = MM + q0 + 16;                 // one past max valid key in tile
  int nchunks = (smax + 31) >> 5;

  for (int ch = 0; ch < nchunks; ++ch) {
    int c0 = ch * 32;
    if (ch + 1 < nchunks) {                // -> global_prefetch_b8
      __builtin_prefetch(kbase + (size_t)(c0 + 32) * DD, 0, 1);
      __builtin_prefetch(vbase + (size_t)(c0 + 32), 0, 1);
    }

    // ---- K fragments and score WMMAs ----
    v16h bk00 = load_bfrag(kbase + (size_t)c0 * DD,             DD, lane);
    v16h bk01 = load_bfrag(kbase + (size_t)c0 * DD + 32,        DD, lane);
    v16h bk10 = load_bfrag(kbase + (size_t)(c0 + 16) * DD,      DD, lane);
    v16h bk11 = load_bfrag(kbase + (size_t)(c0 + 16) * DD + 32, DD, lane);
    // ---- V fragments issued early (consumed after softmax) ----
    v16h bv0 = load_bfrag(vbase + (size_t)(0 * 16) * SS + c0, SS, lane);
    v16h bv1 = load_bfrag(vbase + (size_t)(1 * 16) * SS + c0, SS, lane);
    v16h bv2 = load_bfrag(vbase + (size_t)(2 * 16) * SS + c0, SS, lane);
    v16h bv3 = load_bfrag(vbase + (size_t)(3 * 16) * SS + c0, SS, lane);

    v8f s0 = {}, s1 = {};
    s0 = wmma_f16(aq0, bk00, s0); s0 = wmma_f16(aq1, bk01, s0);
    s1 = wmma_f16(aq0, bk10, s1); s1 = wmma_f16(aq1, bk11, s1);

    bool needmask = (c0 + 31 > MM + q0);   // uniform per wave
#pragma unroll
    for (int r = 0; r < 8; ++r) {
      float v0 = s0[r] * 0.125f;           // 1/sqrt(64)
      float v1 = s1[r] * 0.125f;
      if (needmask) {
        int trow = q0 + r + 8 * half;
        int sk0 = c0 + nl, sk1 = c0 + 16 + nl;
        if (sk0 >= MM && sk0 - MM > trow) v0 = -3.0e38f;
        if (sk1 >= MM && sk1 - MM > trow) v1 = -3.0e38f;
      }
      s0[r] = v0; s1[r] = v1;
    }

    // ---- online softmax (row = reg r + 8*half; reduce across 16 lanes) ----
#pragma unroll
    for (int r = 0; r < 8; ++r) {
      float mx = fmaxf(s0[r], s1[r]);
      mx = fmaxf(mx, __shfl_xor(mx, 1));
      mx = fmaxf(mx, __shfl_xor(mx, 2));
      mx = fmaxf(mx, __shfl_xor(mx, 4));
      mx = fmaxf(mx, __shfl_xor(mx, 8));
      float mnew = fmaxf(mrow[r], mx);
      float corr = __expf(mrow[r] - mnew);
      float p0 = __expf(s0[r] - mnew);
      float p1 = __expf(s1[r] - mnew);
      float sum = p0 + p1;
      sum += __shfl_xor(sum, 1);
      sum += __shfl_xor(sum, 2);
      sum += __shfl_xor(sum, 4);
      sum += __shfl_xor(sum, 8);
      lrow[r] = lrow[r] * corr + sum;
      mrow[r] = mnew;
      o[0][r] *= corr; o[1][r] *= corr; o[2][r] *= corr; o[3][r] *= corr;
      // stash probs (C/D layout -> row-major LDS) for A-fragmentation
      plds[wavib][(r + 8 * half) * 32 + nl]      = (f16)p0;
      plds[wavib][(r + 8 * half) * 32 + nl + 16] = (f16)p1;
    }
    asm volatile("s_wait_dscnt 0" ::: "memory");  // LDS writes visible in-wave
    __builtin_amdgcn_wave_barrier();

    // ---- P as 16x32 A-fragment from LDS ----
    v16h ap;
    {
      const f16* p = &plds[wavib][(lane & 15) * 32 + (lane >> 4) * 16];
      *((v8h*)&ap)       = *(const v8h*)(p);
      *(((v8h*)&ap) + 1) = *(const v8h*)(p + 8);
    }

    // ---- O += P @ V (fragments already in flight) ----
    o[0] = wmma_f16(ap, bv0, o[0]);
    o[1] = wmma_f16(ap, bv1, o[1]);
    o[2] = wmma_f16(ap, bv2, o[2]);
    o[3] = wmma_f16(ap, bv3, o[3]);
  }

  // ---- normalize and write y_att [B,T,C] as f16 ----
#pragma unroll
  for (int j = 0; j < 4; ++j) {
#pragma unroll
    for (int r = 0; r < 8; ++r) {
      int trow = q0 + r + 8 * half;
      float val = o[j][r] / lrow[r];
      yh[((size_t)b * TT + trow) * CC + h * DD + j * 16 + nl] = (f16)val;
    }
  }
}

// ------------------------------ proj GEMM ----------------------------------

__global__ void __launch_bounds__(256, 1)
k_gemm_proj(const f16* __restrict__ ah, const f16* __restrict__ wt,
            const float* __restrict__ bias, float* __restrict__ out) {
  int tid = threadIdx.x;
  int lane = tid & 31;
  int wave = tid >> 5;
  int m0 = blockIdx.y * 256 + wave * 32;
  int n0 = blockIdx.x * 64;
  const f16* abase = ah + (size_t)m0 * CC;
  v8f c[2][4] = {};
  GEMM_CORE(abase, wt, n0, c)

  int half = lane >> 4, nl = lane & 15;
#pragma unroll
  for (int i = 0; i < 2; ++i) {
#pragma unroll
    for (int j = 0; j < 4; ++j) {
      int ncol = n0 + j * 16 + nl;
      float bv = bias[ncol];
#pragma unroll
      for (int r = 0; r < 8; ++r) {
        int m = m0 + i * 16 + r + 8 * half;
        out[(size_t)m * CC + ncol] = c[i][j][r] + bv;
      }
    }
  }
}

// ------------------------------- launcher ----------------------------------

extern "C" void kernel_launch(void* const* d_in, const int* in_sizes, int n_in,
                              void* d_out, int out_size, void* d_ws, size_t ws_size,
                              hipStream_t stream) {
  const float* x     = (const float*)d_in[0];
  const float* em    = (const float*)d_in[1];
  const float* Wattn = (const float*)d_in[2];
  const float* battn = (const float*)d_in[3];
  const float* Wproj = (const float*)d_in[4];
  const float* bproj = (const float*)d_in[5];
  float* out = (float*)d_out;

  // Workspace carve-up (~95 MB, 256B aligned slabs)
  char* ws = (char*)d_ws;
  size_t off = 0;
  auto alloc = [&](size_t bytes) -> void* {
    void* p = ws + off;
    off = (off + bytes + 255) & ~(size_t)255;
    return p;
  };
  f16* xh  = (f16*)alloc((size_t)BB * TT * CC * 2);      // x in f16
  f16* wat = (f16*)alloc((size_t)C3 * CC * 2);           // W_attn^T f16
  f16* wpt = (f16*)alloc((size_t)CC * CC * 2);           // W_proj^T f16
  f16* qh  = (f16*)alloc((size_t)BB * HH * TT * DD * 2); // Q [B,H,T,D]
  f16* kh  = (f16*)alloc((size_t)BB * HH * SS * DD * 2); // K [B,H,S,D]
  f16* vt  = (f16*)alloc((size_t)BB * HH * DD * SS * 2); // V^T [B,H,D,S]
  f16* yh  = (f16*)alloc((size_t)BB * TT * CC * 2);      // attention out f16
  (void)ws_size; (void)in_sizes; (void)n_in; (void)out_size;

  int n;
  n = BB * TT * CC;
  k_convert_x<<<(n + 255) / 256, 256, 0, stream>>>(x, xh, n);
  n = CC * C3;
  k_transpose_w<<<(n + 255) / 256, 256, 0, stream>>>(Wattn, wat, CC, C3);
  n = CC * CC;
  k_transpose_w<<<(n + 255) / 256, 256, 0, stream>>>(Wproj, wpt, CC, CC);
  n = BB * MM * CC;
  k_mem_convert<<<(n + 255) / 256, 256, 0, stream>>>(em, kh, vt);

  k_gemm_qkv<<<dim3(C3 / 64, (BB * TT) / 256), 256, 0, stream>>>(
      xh, wat, battn, qh, kh, vt);

  k_attn<<<dim3((BB * HH * (TT / 16)) / 8), 256, 0, stream>>>(qh, kh, vt, yh);

  k_gemm_proj<<<dim3(CC / 64, (BB * TT) / 256), 256, 0, stream>>>(
      yh, wpt, bproj, out);
}
